// TransformerEncoder_40664750359404
// MI455X (gfx1250) — matrix operand order repaired and checked by reading, not compiled, not run on hardware
//
#include <hip/hip_runtime.h>
#include <hip/hip_bf16.h>

// ---------------- WMMA types ----------------
typedef __attribute__((ext_vector_type(16))) _Float16 v16h;
typedef __attribute__((ext_vector_type(8)))  float    v8f;

union F16x16 { v16h v; uint4 u[2]; _Float16 e[16]; };

__device__ __forceinline__ v8f wmma32(v16h a, v16h b, v8f c) {
  // D = A(16x32) * B(32x16) + C, f32 accumulate
  return __builtin_amdgcn_wmma_f32_16x16x32_f16(false, a, false, b, (short)0, c,
                                                false, false);
}

__device__ __forceinline__ float gelu_f(float x) {
  return 0.5f * x * (1.0f + erff(x * 0.70710678118654752f));
}

// ---------------- model constants ----------------
#define MB   4
#define MS   2048
#define MD   1024
#define MH   16
#define MHD  64
#define MFF  2048
#define MM   (MB * MS)      // 8192 rows
#define MC   10

// ======================================================================
// f32 -> f16 conversion
// ======================================================================
__global__ __launch_bounds__(256) void f32_to_f16_kernel(const float* __restrict__ in,
                                                         _Float16* __restrict__ out,
                                                         size_t n) {
  size_t i = (size_t)blockIdx.x * 256 + threadIdx.x;
  if (i < n) out[i] = (_Float16)in[i];
}

// ======================================================================
// embedding + sinusoidal positional encoding -> f32 residual + f16 copy
// ======================================================================
__global__ __launch_bounds__(256) void embed_pe_kernel(const int* __restrict__ ids,
                                                       const float* __restrict__ emb,
                                                       float* __restrict__ xf,
                                                       _Float16* __restrict__ xh) {
  size_t i = (size_t)blockIdx.x * 256 + threadIdx.x;   // over B*S*D
  if (i >= (size_t)MM * MD) return;
  int d = (int)(i % MD);
  size_t row = i / MD;            // b*S+s
  int s = (int)(row % MS);
  int id = ids[row];
  int half = d >> 1;
  float freq = __expf(-(logf(10000.0f) / (float)MD) * (float)(2 * half));
  float ang = (float)s * freq;
  float pe = (d & 1) ? cosf(ang) : sinf(ang);
  float v = emb[(size_t)id * MD + d] + pe;
  xf[i] = v;
  xh[i] = (_Float16)v;
}

// ======================================================================
// generic WMMA GEMM:  C[M,N] = A[M,K] * W[N,K]^T (+bias)(+gelu)
// A,W row-major f16.  M,N multiples of 128; K multiple of 32.
// block 256 thr (8 waves); block tile 128x128; wave tile 32x64 (2x4 wmma)
// ======================================================================
#define BM 128
#define BN 128
#define BK 32
#define BKP 40   // padded LDS stride (halfs); 80B keeps 16B alignment

__global__ __launch_bounds__(256) void gemm_f16_kernel(
    const _Float16* __restrict__ A, const _Float16* __restrict__ W,
    const float* __restrict__ bias, float* __restrict__ outF,
    _Float16* __restrict__ outH, int M, int N, int K, int act_gelu) {
  __shared__ _Float16 sA[BM][BKP];
  __shared__ _Float16 sB[BN][BKP];

  const int t = threadIdx.x;
  const int lane = t & 31, wave = t >> 5;
  const int wm = wave >> 1;          // 0..3  -> rows wm*32
  const int wn = wave & 1;           // 0..1  -> cols wn*64
  const int m0 = blockIdx.y * BM;
  const int n0 = blockIdx.x * BN;

  v8f acc[2][4] = {};

  const int lr = t >> 1;             // 0..127
  const int lc = (t & 1) * 16;       // 0 or 16

  const int rr = lane & 15;          // A row / B col within 16-tile
  const int kg = lane >> 4;          // lane-half selector

  for (int k0 = 0; k0 < K; k0 += BK) {
    // cooperative global -> LDS (each thread 16 halfs of A and of W)
    const uint4* ga = reinterpret_cast<const uint4*>(&A[(size_t)(m0 + lr) * K + k0 + lc]);
    uint4 a0 = ga[0], a1 = ga[1];
    const uint4* gw = reinterpret_cast<const uint4*>(&W[(size_t)(n0 + lr) * K + k0 + lc]);
    uint4 b0 = gw[0], b1 = gw[1];
    __syncthreads();
    *reinterpret_cast<uint4*>(&sA[lr][lc])     = a0;
    *reinterpret_cast<uint4*>(&sA[lr][lc + 8]) = a1;
    *reinterpret_cast<uint4*>(&sB[lr][lc])     = b0;
    *reinterpret_cast<uint4*>(&sB[lr][lc + 8]) = b1;
    __syncthreads();

    // A fragments (16x32): lane half kg holds K kg*16..kg*16+15
    F16x16 af[2];
#pragma unroll
    for (int i = 0; i < 2; i++) {
      int arow = wm * 32 + i * 16 + rr;
      af[i].u[0] = *reinterpret_cast<const uint4*>(&sA[arow][kg * 16]);
      af[i].u[1] = *reinterpret_cast<const uint4*>(&sA[arow][kg * 16 + 8]);
    }
    // B fragments (32x16): lane half kg holds K kg*16..kg*16+15 of col rr
    F16x16 bf[4];
#pragma unroll
    for (int j = 0; j < 4; j++) {
      int nrow = wn * 64 + j * 16 + rr;
      bf[j].u[0] = *reinterpret_cast<const uint4*>(&sB[nrow][kg * 16]);
      bf[j].u[1] = *reinterpret_cast<const uint4*>(&sB[nrow][kg * 16 + 8]);
    }
#pragma unroll
    for (int i = 0; i < 2; i++)
#pragma unroll
      for (int j = 0; j < 4; j++)
        acc[i][j] = wmma32(af[i].v, bf[j].v, acc[i][j]);
  }

  // epilogue: C layout row=(lane>>4)*8+e, col=lane&15
#pragma unroll
  for (int i = 0; i < 2; i++) {
#pragma unroll
    for (int j = 0; j < 4; j++) {
      int col = n0 + wn * 64 + j * 16 + (lane & 15);
      float bv = bias ? bias[col] : 0.0f;
#pragma unroll
      for (int e = 0; e < 8; e++) {
        int row = m0 + wm * 32 + i * 16 + (lane >> 4) * 8 + e;
        float v = acc[i][j][e] + bv;
        if (act_gelu) v = gelu_f(v);
        size_t idx = (size_t)row * N + col;
        if (outF) outF[idx] = v;
        if (outH) outH[idx] = (_Float16)v;
      }
    }
  }
}

// ======================================================================
// repack qkv [B,S,3D] f16 -> q,k,v [B,H,S,HD] f16 (q pre-scaled 1/8)
// ======================================================================
__global__ __launch_bounds__(256) void repack_qkv_kernel(
    const _Float16* __restrict__ qkv, _Float16* __restrict__ q,
    _Float16* __restrict__ k, _Float16* __restrict__ v) {
  size_t i = (size_t)blockIdx.x * 256 + threadIdx.x;  // over B*S*D
  if (i >= (size_t)MM * MD) return;
  int d = (int)(i % MD);
  size_t row = i / MD;                 // b*S+s
  int b = (int)(row / MS), s = (int)(row % MS);
  int h = d >> 6, dd = d & 63;
  size_t src = row * (size_t)(3 * MD);
  size_t dst = (((size_t)b * MH + h) * MS + s) * MHD + dd;
  q[dst] = (_Float16)((float)qkv[src + d] * 0.125f);   // 1/sqrt(64)
  k[dst] = qkv[src + MD + d];
  v[dst] = qkv[src + 2 * MD + d];
}

// ======================================================================
// flash attention: grid (S/64, B*H), 256 threads (8 waves)
// per block: 64 query rows; loop over key blocks of 64 w/ online softmax
// ======================================================================
#define QT 64
#define KT 64
#define STQ 72   // LDS stride (halfs); 144B row stride, 16B aligned

__global__ __launch_bounds__(256) void flash_attn_kernel(
    const _Float16* __restrict__ Qm, const _Float16* __restrict__ Km,
    const _Float16* __restrict__ Vm, const int* __restrict__ mask,
    _Float16* __restrict__ O) {
  __shared__ _Float16 sQ[QT][STQ];
  __shared__ _Float16 sK[KT][STQ];
  __shared__ _Float16 sVt[MHD][STQ];    // transposed: [d][k]
  __shared__ _Float16 sP[QT][STQ];
  __shared__ float sS[QT][68];
  __shared__ float m_i[QT], l_i[QT], alpha_s[QT];

  const int t = threadIdx.x, lane = t & 31, wave = t >> 5;
  const int qtile = blockIdx.x;
  const int bh = blockIdx.y;
  const int b = bh / MH, h = bh % MH;
  const size_t base = (size_t)bh * MS * MHD;
  const int q0 = qtile * QT;

  // load Q tile (64x64 halfs): 4 threads per row, 16 halfs each
  {
    int r = t >> 2, c = (t & 3) * 16;
    const uint4* g = reinterpret_cast<const uint4*>(&Qm[base + (size_t)(q0 + r) * MHD + c]);
    *reinterpret_cast<uint4*>(&sQ[r][c])     = g[0];
    *reinterpret_cast<uint4*>(&sQ[r][c + 8]) = g[1];
  }
  if (t < QT) { m_i[t] = -1e30f; l_i[t] = 0.0f; }

  v8f acc[2] = {};
  const int wqh = wave >> 2;     // 0..1 : q rows wqh*32..+31
  const int wng = wave & 3;      // 0..3 : 16-col group (k or d)
  const int rr = lane & 15, kg = lane >> 4;

  for (int kb = 0; kb < MS; kb += KT) {
    __syncthreads();   // previous iteration fully consumed sK/sVt/sS/sP
    {
      int r = t >> 2, c = (t & 3) * 16;
      const uint4* g = reinterpret_cast<const uint4*>(&Km[base + (size_t)(kb + r) * MHD + c]);
      *reinterpret_cast<uint4*>(&sK[r][c])     = g[0];
      *reinterpret_cast<uint4*>(&sK[r][c + 8]) = g[1];
      F16x16 vv;
      const uint4* gv = reinterpret_cast<const uint4*>(&Vm[base + (size_t)(kb + r) * MHD + c]);
      vv.u[0] = gv[0]; vv.u[1] = gv[1];
#pragma unroll
      for (int i = 0; i < 16; i++) sVt[c + i][r] = vv.e[i];
    }
    __syncthreads();

    // ---- scores S = Q K^T (q pre-scaled); wave: 2 q-subtiles x 1 k-group
    v8f sfr[2] = {};
#pragma unroll
    for (int ks = 0; ks < MHD; ks += 32) {
      F16x16 bf;
      int brow = wng * 16 + rr;
      bf.u[0] = *reinterpret_cast<const uint4*>(&sK[brow][ks + kg * 16]);
      bf.u[1] = *reinterpret_cast<const uint4*>(&sK[brow][ks + kg * 16 + 8]);
#pragma unroll
      for (int i = 0; i < 2; i++) {
        int arow = wqh * 32 + i * 16 + rr;
        F16x16 af;
        af.u[0] = *reinterpret_cast<const uint4*>(&sQ[arow][ks + kg * 16]);
        af.u[1] = *reinterpret_cast<const uint4*>(&sQ[arow][ks + kg * 16 + 8]);
        sfr[i] = wmma32(af.v, bf.v, sfr[i]);
      }
    }
#pragma unroll
    for (int i = 0; i < 2; i++)
#pragma unroll
      for (int e = 0; e < 8; e++)
        sS[wqh * 32 + i * 16 + (lane >> 4) * 8 + e][wng * 16 + (lane & 15)] = sfr[i][e];
    __syncthreads();

    // ---- online softmax over this key block (one thread per q row)
    if (t < QT) {
      float mo = m_i[t], mloc = mo;
      for (int c = 0; c < KT; c++) {
        float xv = mask[b * MS + kb + c] ? sS[t][c] : -1e30f;
        mloc = fmaxf(mloc, xv);
      }
      float al = __expf(mo - mloc);
      float ls = 0.0f;
      for (int c = 0; c < KT; c++) {
        float xv = mask[b * MS + kb + c] ? sS[t][c] : -1e30f;
        float p = (xv <= -1e29f) ? 0.0f : __expf(xv - mloc);
        sP[t][c] = (_Float16)p;
        ls += p;
      }
      m_i[t] = mloc;
      l_i[t] = l_i[t] * al + ls;
      alpha_s[t] = al;
    }
    __syncthreads();

    // ---- rescale accumulators, then acc += P V
#pragma unroll
    for (int i = 0; i < 2; i++)
#pragma unroll
      for (int e = 0; e < 8; e++)
        acc[i][e] *= alpha_s[wqh * 32 + i * 16 + (lane >> 4) * 8 + e];

#pragma unroll
    for (int ks = 0; ks < KT; ks += 32) {
      F16x16 bf;
      int drow = wng * 16 + rr;
      bf.u[0] = *reinterpret_cast<const uint4*>(&sVt[drow][ks + kg * 16]);
      bf.u[1] = *reinterpret_cast<const uint4*>(&sVt[drow][ks + kg * 16 + 8]);
#pragma unroll
      for (int i = 0; i < 2; i++) {
        int arow = wqh * 32 + i * 16 + rr;
        F16x16 af;
        af.u[0] = *reinterpret_cast<const uint4*>(&sP[arow][ks + kg * 16]);
        af.u[1] = *reinterpret_cast<const uint4*>(&sP[arow][ks + kg * 16 + 8]);
        acc[i] = wmma32(af.v, bf.v, acc[i]);
      }
    }
  }

  // ---- write O[b][s][h*64+d] = acc / l
#pragma unroll
  for (int i = 0; i < 2; i++) {
#pragma unroll
    for (int e = 0; e < 8; e++) {
      int qr = wqh * 32 + i * 16 + (lane >> 4) * 8 + e;
      int d = wng * 16 + (lane & 15);
      float v = acc[i][e] / l_i[qr];
      O[((size_t)b * MS + q0 + qr) * MD + h * MHD + d] = (_Float16)v;
    }
  }
}

// ======================================================================
// fused residual add + LayerNorm (per row of D=1024), f32 (+optional f16) out
// ======================================================================
__global__ __launch_bounds__(256) void add_ln_kernel(
    const float* __restrict__ a, const float* __restrict__ resid,
    const float* __restrict__ g, const float* __restrict__ bta,
    float* __restrict__ outF, _Float16* __restrict__ outH) {
  __shared__ float red[256];
  const int row = blockIdx.x, t = threadIdx.x;
  float vals[4];
  float s = 0.0f;
#pragma unroll
  for (int j = 0; j < 4; j++) {
    int d = t + j * 256;
    float v = a[(size_t)row * MD + d] + resid[(size_t)row * MD + d];
    vals[j] = v; s += v;
  }
  red[t] = s; __syncthreads();
  for (int off = 128; off > 0; off >>= 1) { if (t < off) red[t] += red[t + off]; __syncthreads(); }
  float mu = red[0] * (1.0f / MD);
  __syncthreads();
  float s2 = 0.0f;
#pragma unroll
  for (int j = 0; j < 4; j++) { float dv = vals[j] - mu; s2 += dv * dv; }
  red[t] = s2; __syncthreads();
  for (int off = 128; off > 0; off >>= 1) { if (t < off) red[t] += red[t + off]; __syncthreads(); }
  float inv = rsqrtf(red[0] * (1.0f / MD) + 1e-5f);
#pragma unroll
  for (int j = 0; j < 4; j++) {
    int d = t + j * 256;
    float o = (vals[j] - mu) * inv * g[d] + bta[d];
    outF[(size_t)row * MD + d] = o;
    if (outH) outH[(size_t)row * MD + d] = (_Float16)o;
  }
}

// ======================================================================
// masked max-pool over S: grid B*D blocks
// ======================================================================
__global__ __launch_bounds__(256) void maxpool_kernel(const float* __restrict__ x,
                                                      const int* __restrict__ mask,
                                                      float* __restrict__ out) {
  __shared__ float red[256];
  int b = blockIdx.x / MD, d = blockIdx.x % MD, t = threadIdx.x;
  float m = -1e30f;
  for (int s = t; s < MS; s += 256)
    if (mask[b * MS + s]) m = fmaxf(m, x[((size_t)b * MS + s) * MD + d]);
  red[t] = m; __syncthreads();
  for (int off = 128; off > 0; off >>= 1) { if (t < off) red[t] = fmaxf(red[t], red[t + off]); __syncthreads(); }
  if (t == 0) out[(size_t)b * MD + d] = red[0];
}

// ======================================================================
// small head: out[b][j] = act(dot(x[b], w[j]) + bias[j]); grid B*N blocks
// ======================================================================
__global__ __launch_bounds__(256) void head_dot_kernel(
    const float* __restrict__ xin, const float* __restrict__ w,
    const float* __restrict__ bias, float* __restrict__ out,
    int N, int Kd, int act_gelu) {
  __shared__ float red[256];
  int b = blockIdx.x / N, j = blockIdx.x % N, t = threadIdx.x;
  float s = 0.0f;
  for (int k = t; k < Kd; k += 256) s += xin[(size_t)b * Kd + k] * w[(size_t)j * Kd + k];
  red[t] = s; __syncthreads();
  for (int off = 128; off > 0; off >>= 1) { if (t < off) red[t] += red[t + off]; __syncthreads(); }
  if (t == 0) {
    float v = red[0] + bias[j];
    if (act_gelu) v = gelu_f(v);
    out[(size_t)b * N + j] = v;
  }
}

// ======================================================================
extern "C" void kernel_launch(void* const* d_in, const int* in_sizes, int n_in,
                              void* d_out, int out_size, void* d_ws, size_t ws_size,
                              hipStream_t stream) {
  (void)in_sizes; (void)n_in; (void)out_size; (void)ws_size;
  const int*   x_ids  = (const int*)  d_in[0];
  const int*   maskp  = (const int*)  d_in[1];
  const float* emb    = (const float*)d_in[2];
  const float* qkv_w  = (const float*)d_in[3];
  const float* fc_w   = (const float*)d_in[4];
  const float* fc_b   = (const float*)d_in[5];
  const float* ln1_g  = (const float*)d_in[6];
  const float* ln1_b  = (const float*)d_in[7];
  const float* ffn_w1 = (const float*)d_in[8];
  const float* ffn_b1 = (const float*)d_in[9];
  const float* ffn_w2 = (const float*)d_in[10];
  const float* ffn_b2 = (const float*)d_in[11];
  const float* ln2_g  = (const float*)d_in[12];
  const float* ln2_b  = (const float*)d_in[13];
  const float* pr_w1  = (const float*)d_in[14];
  const float* pr_b1  = (const float*)d_in[15];
  const float* pr_w2  = (const float*)d_in[16];
  const float* pr_b2  = (const float*)d_in[17];
  float* out = (float*)d_out;

  char* ws = (char*)d_ws;
  size_t off = 0;
  auto alloc = [&](size_t bytes) -> char* {
    char* p = ws + off;
    off += (bytes + 255) & ~(size_t)255;
    return p;
  };

  const size_t ND = (size_t)MM * MD;     // 8.4M activations
  _Float16* wq_h  = (_Float16*)alloc((size_t)3 * MD * MD * 2);
  _Float16* wfc_h = (_Float16*)alloc((size_t)MD * MD * 2);
  _Float16* wf1_h = (_Float16*)alloc((size_t)MFF * MD * 2);
  _Float16* wf2_h = (_Float16*)alloc((size_t)MD * MFF * 2);
  char* R1 = alloc(ND * 4);              // x_f32, later ln2_f32
  char* R2 = alloc(ND * 2);              // x_h, later o_h
  char* R3 = alloc(ND * 3 * 2);          // qkv_h, later ffn hidden f16
  char* R4 = alloc(ND * 3 * 2);          // q/k/v, later ln1_f32 + ln1_h
  char* R5 = alloc(ND * 4);              // fcout f32, later ffn2out f32
  float* pooled = (float*)alloc((size_t)MB * MD * 4);
  float* h1     = (float*)alloc((size_t)MB * MD * 4);

  float*    x_f32 = (float*)R1;
  _Float16* x_h   = (_Float16*)R2;
  _Float16* qkv_h = (_Float16*)R3;
  _Float16* q_h   = (_Float16*)R4;
  _Float16* k_h   = q_h + ND;
  _Float16* v_h   = k_h + ND;
  _Float16* o_h   = (_Float16*)R2;       // reuse after QKV GEMM
  float*    fcout = (float*)R5;
  float*    ln1_f = (float*)R4;          // reuse after attention
  _Float16* ln1_h = (_Float16*)(R4 + ND * 4);
  _Float16* ffnh  = (_Float16*)R3;       // reuse after repack
  float*    ffn2o = (float*)R5;          // reuse after LN1
  float*    ln2_f = (float*)R1;          // reuse after LN1

  const int eltBlocks = (int)((ND + 255) / 256);

  // 1) weights -> f16
  f32_to_f16_kernel<<<(3 * MD * MD + 255) / 256, 256, 0, stream>>>(qkv_w, wq_h, (size_t)3 * MD * MD);
  f32_to_f16_kernel<<<(MD * MD + 255) / 256, 256, 0, stream>>>(fc_w, wfc_h, (size_t)MD * MD);
  f32_to_f16_kernel<<<(MFF * MD + 255) / 256, 256, 0, stream>>>(ffn_w1, wf1_h, (size_t)MFF * MD);
  f32_to_f16_kernel<<<(MD * MFF + 255) / 256, 256, 0, stream>>>(ffn_w2, wf2_h, (size_t)MD * MFF);

  // 2) embedding + PE
  embed_pe_kernel<<<eltBlocks, 256, 0, stream>>>(x_ids, emb, x_f32, x_h);

  // 3) QKV projection (no bias)
  gemm_f16_kernel<<<dim3((3 * MD) / BN, MM / BM), 256, 0, stream>>>(
      x_h, wq_h, nullptr, nullptr, qkv_h, MM, 3 * MD, MD, 0);

  // 4) repack to [B,H,S,HD] (q scaled by 1/sqrt(HD))
  repack_qkv_kernel<<<eltBlocks, 256, 0, stream>>>(qkv_h, q_h, k_h, v_h);

  // 5) flash attention -> o_h [B,S,D]
  flash_attn_kernel<<<dim3(MS / QT, MB * MH), 256, 0, stream>>>(q_h, k_h, v_h, maskp, o_h);

  // 6) output projection (+fc_b), f32
  gemm_f16_kernel<<<dim3(MD / BN, MM / BM), 256, 0, stream>>>(
      o_h, wfc_h, fc_b, fcout, nullptr, MM, MD, MD, 0);

  // 7) LN1(fcout + residual)
  add_ln_kernel<<<MM, 256, 0, stream>>>(fcout, x_f32, ln1_g, ln1_b, ln1_f, ln1_h);

  // 8) FFN1 (+bias, gelu) -> f16 hidden
  gemm_f16_kernel<<<dim3(MFF / BN, MM / BM), 256, 0, stream>>>(
      ln1_h, wf1_h, ffn_b1, nullptr, ffnh, MM, MFF, MD, 1);

  // 9) FFN2 (+bias) -> f32
  gemm_f16_kernel<<<dim3(MD / BN, MM / BM), 256, 0, stream>>>(
      ffnh, wf2_h, ffn_b2, ffn2o, nullptr, MM, MD, MFF, 0);

  // 10) LN2(ffn2 + ln1)
  add_ln_kernel<<<MM, 256, 0, stream>>>(ffn2o, ln1_f, ln2_g, ln2_b, ln2_f, nullptr);

  // 11) masked max-pool over S
  maxpool_kernel<<<MB * MD, 256, 0, stream>>>(ln2_f, maskp, pooled);

  // 12) head
  head_dot_kernel<<<MB * MD, 256, 0, stream>>>(pooled, pr_w1, pr_b1, h1, MD, MD, 1);
  head_dot_kernel<<<MB * MC, 256, 0, stream>>>(h1, pr_w2, pr_b2, out, MC, MD, 0);
}
